// ExpertsChooseMaskedRouter_56487409877320
// MI455X (gfx1250) — matrix operand order repaired and checked by reading, not compiled, not run on hardware
//
#include <hip/hip_runtime.h>
#include <stdint.h>

typedef __attribute__((ext_vector_type(16))) _Float16 v16h;
typedef __attribute__((ext_vector_type(8)))  float    v8f;
typedef __attribute__((ext_vector_type(4)))  float    v4f;

namespace {
constexpr int kB = 4;
constexpr int kT = 4096;
constexpr int kD = 768;
constexpr int kE = 8;
constexpr int kNTok = kB * kT;     // 16384 tokens
constexpr int kWavesPerBlock = 8;  // 256 threads, one 16-token tile per wave
}

// ---------------------------------------------------------------------------
// K0: zero-fill the 1 GiB output with non-temporal 16B stores. The output is
// written once and never re-read within the kernel (1 GiB >> 192 MB L2), so
// NT streaming stores are the right cache policy for the dominant traffic.
// ---------------------------------------------------------------------------
__global__ void zero_fill_kernel(v4f* __restrict__ out, size_t n4) {
  size_t i = (size_t)blockIdx.x * blockDim.x + threadIdx.x;
  size_t stride = (size_t)gridDim.x * blockDim.x;
  v4f z = {0.f, 0.f, 0.f, 0.f};
  for (; i < n4; i += stride) __builtin_nontemporal_store(z, &out[i]);
}

// ---------------------------------------------------------------------------
// K1: logits via v_wmma_f32_16x16x32_f16. One wave = 16 tokens x 16 experts
// (experts 8..15 are zero rows of a zero-padded f16 W tile held in LDS, so
// the B-fragment load is an unconditional ds_load -- no exec divergence in
// the K-loop). x is streamed with non-temporal loads. Softmax over E=8 in
// f32 per token; writes pr[b*E+e][t] (expert-major probs) to workspace.
// ---------------------------------------------------------------------------
__global__ __launch_bounds__(256)
void router_logits_softmax_kernel(const float* __restrict__ x,
                                  const float* __restrict__ W,
                                  float* __restrict__ pr) {
  __shared__ __align__(32) _Float16 wsh[16 * kD];        // 24 KB, rows 8..15 = 0
  __shared__ float csh[kWavesPerBlock][16][17];          // C staging, padded

  // ---- cooperative fill of the zero-padded f16 W tile --------------------
  {
    uint32_t* wsh32 = (uint32_t*)wsh;
    constexpr int kPairsPerRow = kD / 2;                 // 384 u32 per row
    for (int i = threadIdx.x; i < 16 * kPairsPerRow; i += 256) {
      const int e  = i / kPairsPerRow;
      const int kk = (i - e * kPairsPerRow) * 2;
      uint32_t packed = 0u;
      if (e < kE) {
        _Float16 h0 = (_Float16)W[(size_t)e * kD + kk];
        _Float16 h1 = (_Float16)W[(size_t)e * kD + kk + 1];
        uint16_t u0, u1;
        __builtin_memcpy(&u0, &h0, 2);
        __builtin_memcpy(&u1, &h1, 2);
        packed = (uint32_t)u0 | ((uint32_t)u1 << 16);
      }
      wsh32[i] = packed;
    }
  }
  __syncthreads();

  const int lane = threadIdx.x & 31;
  const int wave = threadIdx.x >> 5;
  const int tile = blockIdx.x * kWavesPerBlock + wave;
  const int tok0 = tile * 16;
  const int nidx = lane & 15;       // A: token row;  B/C: expert column
  const bool hi  = lane >= 16;
  const int koffA = hi ? 16 : 0;    // this code's A k order: lanes 16-31 hold K=16..31
  const int koffB = hi ? 16 : 0;    // this code's B k order: lanes 16-31 hold K=16..31

  const float* __restrict__ xrow = x + (size_t)(tok0 + nidx) * kD;
  const _Float16* __restrict__ wlds = wsh + (size_t)nidx * kD;

  v8f c = {};
  for (int k0 = 0; k0 < kD; k0 += 32) {
    // A fragment: halves 0..15 = x[tok, k0+koffA..+15]
    v16h a;
    {
      const v4f* p0 = (const v4f*)(xrow + k0 + koffA);
      const v4f* p1 = p0 + 2;
      v4f f0 = __builtin_nontemporal_load(p0);
      v4f f1 = __builtin_nontemporal_load(p0 + 1);
      v4f f2 = __builtin_nontemporal_load(p1);
      v4f f3 = __builtin_nontemporal_load(p1 + 1);
#pragma unroll
      for (int i = 0; i < 4; ++i) {
        a[i]      = (_Float16)f0[i];
        a[4 + i]  = (_Float16)f1[i];
        a[8 + i]  = (_Float16)f2[i];
        a[12 + i] = (_Float16)f3[i];
      }
    }
    // B fragment: 16 consecutive K halves of (zero-padded) expert row, LDS
    v16h b = *(const v16h*)(wlds + k0 + koffB);
    c = __builtin_amdgcn_wmma_f32_16x16x32_f16(false, a, false, b,
                                               (short)0, c, false, false);
  }

  // C layout: lanes 0-15 -> N=lane, M=r; lanes 16-31 -> N=lane-16, M=r+8
#pragma unroll
  for (int r = 0; r < 8; ++r) {
    int tl = hi ? (r + 8) : r;
    csh[wave][tl][nidx] = c[r];
  }
  __syncthreads();

  // One lane per token: softmax over E=8 in f32, write expert-major probs.
  if (lane < 16) {
    const int tok = tok0 + lane;
    float v[kE];
    float m = -3.4e38f;
#pragma unroll
    for (int e = 0; e < kE; ++e) { v[e] = csh[wave][lane][e]; m = fmaxf(m, v[e]); }
    float s = 0.f;
#pragma unroll
    for (int e = 0; e < kE; ++e) { v[e] = __expf(v[e] - m); s += v[e]; }
    const float inv = 1.f / s;
    const int bb = tok / kT;
    const int tt = tok - bb * kT;
#pragma unroll
    for (int e = 0; e < kE; ++e)
      pr[(size_t)(bb * kE + e) * kT + tt] = v[e] * inv;
  }
}

// ---------------------------------------------------------------------------
// K2: for one (b,e) row (LDS-resident, 16 KB), compute each token's descending
// rank with stable index tiebreak (matches lax.top_k), and scatter the 1-hot
// dispatch entry + combine weight into the zero-filled output.
// ---------------------------------------------------------------------------
__global__ __launch_bounds__(256)
void rank_scatter_kernel(const float* __restrict__ pr,
                         const int* __restrict__ cptr,
                         float* __restrict__ dispatch,
                         float* __restrict__ combine,
                         int C) {
  __shared__ float p_sh[kT];
  constexpr int kChunks = kT / 256;            // 16 token-chunks per (b,e) row
  const int be    = blockIdx.x / kChunks;
  const int chunk = blockIdx.x % kChunks;
  const int bb = be / kE;
  const int ee = be % kE;

  const float* __restrict__ row = pr + (size_t)be * kT;
  for (int j = threadIdx.x; j < kT; j += 256) p_sh[j] = row[j];
  __syncthreads();

  const int t = chunk * 256 + threadIdx.x;
  const float p = p_sh[t];
  int rank = 0;
  for (int j = 0; j < kT; ++j) {
    const float q = p_sh[j];
    rank += (q > p) || (q == p && j < t);
  }

  const int cap = *cptr;  // capacity c (== C here)
  if (rank < cap && rank < C) {
    const size_t idx = (((size_t)bb * kT + t) * kE + ee) * (size_t)C + rank;
    dispatch[idx] = 1.0f;
    combine[idx]  = p;
  }
}

// ---------------------------------------------------------------------------
extern "C" void kernel_launch(void* const* d_in, const int* in_sizes, int n_in,
                              void* d_out, int out_size, void* d_ws, size_t ws_size,
                              hipStream_t stream) {
  const float* x    = (const float*)d_in[0];  // [B,T,D] f32
  const float* W    = (const float*)d_in[1];  // [E,D]   f32
  const int*   cptr = (const int*)d_in[2];    // scalar capacity

  float* out = (float*)d_out;                 // dispatch | combine, concatenated
  float* pr  = (float*)d_ws;                  // [B*E, T] f32 probs (512 KB)

  const size_t half = (size_t)out_size / 2;               // elements per array
  const int C = (int)(half / ((size_t)kB * kT * kE));     // capacity dim
  float* dispatch = out;
  float* combine  = out + half;

  // K0: zero the full 1 GiB output (store-BW bound: ~46 us floor at 23.3 TB/s)
  zero_fill_kernel<<<2048, 256, 0, stream>>>((v4f*)out, (size_t)out_size / 4);

  // K1: WMMA logits + softmax -> pr
  router_logits_softmax_kernel<<<kNTok / 16 / kWavesPerBlock, 256, 0, stream>>>(
      x, W, pr);

  // K2: rank + sparse scatter of the 65,536 nonzero entries
  rank_scatter_kernel<<<kB * kE * (kT / 256), 256, 0, stream>>>(
      pr, cptr, dispatch, combine, C);
}
